// cofs_network_74861279969496
// MI455X (gfx1250) — matrix operand order repaired and checked by reading, not compiled, not run on hardware
//
#include <hip/hip_runtime.h>
#include <hip/hip_bf16.h>
#include <math.h>

// ---------------- model constants ----------------
#define BATCH   32
#define AOBJ    8
#define OMAXC   12
#define LTOK    96          // A*OMAX
#define DMODEL  512
#define NHEAD   8
#define DHEAD   64
#define DFFN    2048
#define LE      99          // encoder sequence length (L+3)
#define LD      97          // decoder sequence length (L+1)
#define ROWS_E  (BATCH*LE)  // 3168
#define ROWS_D  (BATCH*LD)  // 3104
#define NEGINF  (-1.0e9f)
#define TB      256
#define WPB     8           // waves per block (wave32)

typedef __attribute__((ext_vector_type(16))) __bf16 v16bf;
typedef __attribute__((ext_vector_type(8)))  float  v8f;
typedef __attribute__((ext_vector_type(4)))  int    v4i_;

// ---- CDNA5 async global->LDS path (guarded; fallback = load + ds_store) ----
#if defined(__has_builtin)
#  if __has_builtin(__builtin_amdgcn_global_load_async_to_lds_b128) && \
      __has_builtin(__builtin_amdgcn_s_wait_asynccnt)
#    define USE_ASYNC_LDS 1
#  endif
#endif
#ifndef USE_ASYNC_LDS
#  define USE_ASYNC_LDS 0
#endif

typedef __attribute__((address_space(1))) v4i_ gv4i;
typedef __attribute__((address_space(3))) v4i_ lv4i;

__device__ __forceinline__ void cp16_g2l(void* lds, const void* gmem) {
#if USE_ASYNC_LDS
  __builtin_amdgcn_global_load_async_to_lds_b128((gv4i*)gmem, (lv4i*)lds, 0, 0);
#else
  *(float4*)lds = *(const float4*)gmem;
#endif
}
template <int N>
__device__ __forceinline__ void cp_wait() {
#if USE_ASYNC_LDS
  __builtin_amdgcn_s_wait_asynccnt(N);
#endif
}

__device__ __forceinline__ __bf16 f2bf(float x) {
  unsigned u = __builtin_bit_cast(unsigned, x);
  unsigned short h = (unsigned short)(u >> 16);   // truncate to bf16
  return __builtin_bit_cast(__bf16, h);
}

__device__ __forceinline__ float sinembed(float v, int d) {
  int i = d & 255;
  float inv = __expf(-9.210340371976184f * (float)i * (1.0f / 256.0f));
  float ang = v * inv;
  return (d < 256) ? __sinf(ang) : __cosf(ang);
}

// =====================================================================
// Staged WMMA GEMM: C[M,N] = A[M,K] @ W[K,N] (+bias)(+resid)(relu)
// Block = 256 threads = 8 waves arranged 2(M) x 4(N); block tile 32x64.
// K-loop in steps of 32 with double-buffered LDS staging via async copies.
// Requires M%32==0, N%64==0, K%32==0.
// =====================================================================
#define BM 32
#define BN 64
#define BK 32

__global__ void gemm_bf16_wmma_staged(const float* __restrict__ A, int lda,
                                      const float* __restrict__ W, int ldb,
                                      const float* __restrict__ bias,
                                      const float* __restrict__ resid,
                                      float* __restrict__ C, int ldc,
                                      int M, int N, int K, int doRelu)
{
  __shared__ float As[2][BM * BK];   // [m][k]  4 KB per buffer
  __shared__ float Bs[2][BK * BN];   // [k][n]  8 KB per buffer

  const int tid  = threadIdx.x;
  const int lane = tid & 31;
  const int wave = tid >> 5;     // 0..7
  const int mw   = wave >> 2;    // 0..1
  const int nw   = wave & 3;     // 0..3

  const int nblk = N / BN;
  const int m0b  = (blockIdx.x / nblk) * BM;
  const int n0b  = (blockIdx.x % nblk) * BN;

  // staging map: A -> 256 lanes x 16B (one b128 each); B -> 256 lanes x 2 b128
  const int ar  = tid >> 3;          // 0..31 A row
  const int ac  = (tid & 7) * 4;     // A col quad
  const int bk  = tid >> 3;          // 0..31 B k row
  const int bn8 = (tid & 7) * 8;     // B col oct

  const float* gA = A + (size_t)(m0b + ar) * lda + ac;
  const float* gB = W + (size_t)bk * ldb + n0b + bn8;

  // fragment positions (CDNA5 ISA 7.12.2 bf16 16x16x32 layouts)
  const int frow = 16 * mw + (lane & 15);
  const int kbA  = (lane >> 4) << 3;     // 0 / 8
  const int fcol = 16 * nw + (lane & 15);
  const int kgB  = (lane >> 4) << 3;     // 0 / 8

  v8f acc = {0.f, 0.f, 0.f, 0.f, 0.f, 0.f, 0.f, 0.f};

  const int KT = K / BK;
  // stage 0
  cp16_g2l(&As[0][ar * BK + ac], gA);
  cp16_g2l(&Bs[0][bk * BN + bn8], gB);
  cp16_g2l(&Bs[0][bk * BN + bn8 + 4], gB + 4);

  for (int kt = 0; kt < KT; ++kt) {
    const int cur = kt & 1;
    if (kt + 1 < KT) {                       // issue next stage
      const int nxt = cur ^ 1;
      const float* gA1 = gA + (size_t)(kt + 1) * BK;
      const float* gB1 = gB + (size_t)(kt + 1) * BK * ldb;
      cp16_g2l(&As[nxt][ar * BK + ac], gA1);
      cp16_g2l(&Bs[nxt][bk * BN + bn8], gB1);
      cp16_g2l(&Bs[nxt][bk * BN + bn8 + 4], gB1 + 4);
      cp_wait<3>();                          // current stage complete (in-order)
    } else {
      cp_wait<0>();
    }
    __syncthreads();                         // staging visible to all waves

    // A fragment: 4x ds_load_b128 from LDS, convert to bf16
    const float* ap = &As[cur][frow * BK + kbA];
    float4 a0 = *(const float4*)(ap);
    float4 a1 = *(const float4*)(ap + 4);
    float4 a2 = *(const float4*)(ap + 16);
    float4 a3 = *(const float4*)(ap + 20);
    v16bf af;
    af[0]  = f2bf(a0.x); af[1]  = f2bf(a0.y); af[2]  = f2bf(a0.z); af[3]  = f2bf(a0.w);
    af[4]  = f2bf(a1.x); af[5]  = f2bf(a1.y); af[6]  = f2bf(a1.z); af[7]  = f2bf(a1.w);
    af[8]  = f2bf(a2.x); af[9]  = f2bf(a2.y); af[10] = f2bf(a2.z); af[11] = f2bf(a2.w);
    af[12] = f2bf(a3.x); af[13] = f2bf(a3.y); af[14] = f2bf(a3.z); af[15] = f2bf(a3.w);

    v16bf bw;
#pragma unroll
    for (int e = 0; e < 16; ++e)
      bw[e] = f2bf(Bs[cur][(kgB + (e & 7) + ((e >> 3) << 4)) * BN + fcol]);

    acc = __builtin_amdgcn_wmma_f32_16x16x32_bf16(false, af, false, bw,
                                                  (short)0, acc, false, false);
    __syncthreads();                         // all reads done before overwrite
  }

  const int col = n0b + fcol;
  const float bia = bias ? bias[col] : 0.f;
  const int rbase = m0b + 16 * mw + ((lane >> 4) << 3);
#pragma unroll
  for (int r = 0; r < 8; ++r) {
    const int rw = rbase + r;
    float v = acc[r] + bia;
    if (resid)  v += resid[(size_t)rw * ldc + col];
    if (doRelu) v = fmaxf(v, 0.f);
    C[(size_t)rw * ldc + col] = v;
  }
}

// --------- simple one-wave-per-16x16-tile WMMA GEMM (for N==32 logit heads) ---------
__global__ void gemm_bf16_wmma_small(const float* __restrict__ A, int lda,
                                     const float* __restrict__ W, int ldb,
                                     const float* __restrict__ bias,
                                     float* __restrict__ C, int ldc,
                                     int M, int N, int K)
{
  const int lane   = threadIdx.x & 31;
  const int wave   = threadIdx.x >> 5;
  const int ntiles = N >> 4;
  const int mtiles = M >> 4;
  int tile = blockIdx.x * (blockDim.x >> 5) + wave;
  if (tile >= ntiles * mtiles) return;
  const int m0 = (tile / ntiles) << 4;
  const int n0 = (tile % ntiles) << 4;

  const int am  = m0 + (lane & 15);
  const int kbA = (lane >> 4) << 3;
  const int bn  = n0 + (lane & 15);
  const int kgB = (lane >> 4) << 3;

  const float* Arow = A + (size_t)am * lda;
  v8f acc = {0.f, 0.f, 0.f, 0.f, 0.f, 0.f, 0.f, 0.f};

  for (int k0 = 0; k0 < K; k0 += 32) {
    v16bf af, bw;
    const float4* pa = (const float4*)(Arow + k0 + kbA);
    float4 a0 = pa[0], a1 = pa[1];
    const float4* pa2 = (const float4*)(Arow + k0 + kbA + 16);
    float4 a2 = pa2[0], a3 = pa2[1];
    af[0]  = f2bf(a0.x); af[1]  = f2bf(a0.y); af[2]  = f2bf(a0.z); af[3]  = f2bf(a0.w);
    af[4]  = f2bf(a1.x); af[5]  = f2bf(a1.y); af[6]  = f2bf(a1.z); af[7]  = f2bf(a1.w);
    af[8]  = f2bf(a2.x); af[9]  = f2bf(a2.y); af[10] = f2bf(a2.z); af[11] = f2bf(a2.w);
    af[12] = f2bf(a3.x); af[13] = f2bf(a3.y); af[14] = f2bf(a3.z); af[15] = f2bf(a3.w);
    const float* pb = W + (size_t)(k0 + kgB) * ldb + bn;
#pragma unroll
    for (int e = 0; e < 16; ++e) bw[e] = f2bf(pb[(size_t)((e & 7) + ((e >> 3) << 4)) * ldb]);
    acc = __builtin_amdgcn_wmma_f32_16x16x32_bf16(false, af, false, bw,
                                                  (short)0, acc, false, false);
  }

  const int col = n0 + (lane & 15);
  const float bia = bias ? bias[col] : 0.f;
#pragma unroll
  for (int r = 0; r < 8; ++r) {
    const int rw = m0 + r + ((lane >> 4) << 3);
    C[(size_t)rw * ldc + col] = acc[r] + bia;
  }
}

// ---------------- boundary CNN ----------------
__global__ void conv3x3_s2_relu(const float* __restrict__ in, const float* __restrict__ w,
                                const float* __restrict__ bias, float* __restrict__ out,
                                int Bn, int IC, int IH, int IW, int OC)
{
  int OH = IH >> 1, OW = IW >> 1;
  int idx = blockIdx.x * blockDim.x + threadIdx.x;
  int total = Bn * OC * OH * OW;
  if (idx >= total) return;
  int ox = idx % OW; int t = idx / OW;
  int oy = t % OH;  t /= OH;
  int oc = t % OC;  int b = t / OC;
  float acc = bias[oc];
  for (int ic = 0; ic < IC; ++ic)
    for (int ky = 0; ky < 3; ++ky) {
      int iy = 2 * oy + ky; if (iy >= IH) continue;
      for (int kx = 0; kx < 3; ++kx) {
        int ix = 2 * ox + kx; if (ix >= IW) continue;
        acc += in[((size_t)(b * IC + ic) * IH + iy) * IW + ix] *
               w[((oc * IC + ic) * 3 + ky) * 3 + kx];
      }
    }
  out[idx] = fmaxf(acc, 0.f);
}

__global__ void global_pool_16x16(const float* __restrict__ in, float* __restrict__ pooled)
{
  int idx = blockIdx.x * blockDim.x + threadIdx.x;   // b*64 + c
  if (idx >= BATCH * 64) return;
  const float* p = in + (size_t)idx * 256;
  float s = 0.f;
  for (int i = 0; i < 256; ++i) s += p[i];
  pooled[idx] = s * (1.0f / 256.0f);
}

__global__ void fc64_512(const float* __restrict__ pooled, const float* __restrict__ fw,
                         const float* __restrict__ fb, float* __restrict__ out)
{
  int idx = blockIdx.x * blockDim.x + threadIdx.x;   // b*512 + d
  if (idx >= BATCH * DMODEL) return;
  int d = idx & 511; int b = idx >> 9;
  float acc = fb[d];
  for (int c = 0; c < 64; ++c) acc += pooled[b * 64 + c] * fw[c * DMODEL + d];
  out[idx] = acc;
}

// ---------------- sequence assembly ----------------
__global__ void build_enc_seq(const float* __restrict__ seq, const int* __restrict__ sl_,
                              const float* __restrict__ lvec, float* __restrict__ out)
{
  int idx = blockIdx.x * blockDim.x + threadIdx.x;
  if (idx >= ROWS_E * DMODEL) return;
  int d = idx & 511;
  int t = (idx >> 9) % LE;
  int b = idx / (DMODEL * LE);
  int sl = sl_[b]; if (sl > OMAXC) sl = OMAXC;
  float val;
  if (t == 0)       val = sinembed(1.0f, d);                 // SOS
  else if (t == 1)  val = lvec[b * DMODEL + d];              // layout token
  else if (t < 2 + LTOK) {
    int tt = t - 2;
    float m = (tt < AOBJ * sl) ? 1.f : 0.f;
    val = (sinembed(seq[b * LTOK + tt], d) +
           sinembed((float)(tt & 7), d) +
           sinembed((float)(tt >> 3), d)) * m;
  } else val = 0.f;
  if (t == AOBJ * sl + 2) val = sinembed(2.0f, d);           // EOS overwrite
  out[idx] = val;
}

__global__ void build_dec_seq(const float* __restrict__ last, const int* __restrict__ ll_,
                              float* __restrict__ out)
{
  int idx = blockIdx.x * blockDim.x + threadIdx.x;
  if (idx >= ROWS_D * DMODEL) return;
  int d = idx & 511;
  int t = (idx >> 9) % LD;
  int b = idx / (DMODEL * LD);
  float val;
  if (t == 0) val = sinembed(1.0f, d);                       // SOS
  else {
    int tt = t - 1;
    int ll = ll_[b];
    float m = (tt < AOBJ * ll) ? 1.f : 0.f;
    val = (sinembed(last[b * LTOK + tt], d) + sinembed((float)tt, d)) * m;
  }
  out[idx] = val;
}

// ---------------- attention ----------------
__global__ void attn_scores(const float* __restrict__ Q, const float* __restrict__ Kb,
                            float* __restrict__ S, const int* __restrict__ sl_,
                            const int* __restrict__ ll_, int Lq, int Lk, int maskType)
{
  int idx = blockIdx.x * blockDim.x + threadIdx.x;
  int total = BATCH * NHEAD * Lq * Lk;
  if (idx >= total) return;
  int k = idx % Lk; int t = idx / Lk;
  int q = t % Lq;  t /= Lq;
  int h = t % NHEAD; int b = t / NHEAD;
  bool valid;
  if (maskType == 1) {
    int tl = 1 + AOBJ * ll_[b];
    valid = (q < tl) && (k < tl) && (k <= q);
  } else {
    int sl = sl_[b]; if (sl > OMAXC) sl = OMAXC;
    valid = k < (3 + AOBJ * sl);
  }
  float s = NEGINF;
  if (valid) {
    const float* qp = Q + ((size_t)(b * Lq + q)) * DMODEL + h * DHEAD;
    const float* kp = Kb + ((size_t)(b * Lk + k)) * DMODEL + h * DHEAD;
    float acc = 0.f;
#pragma unroll 8
    for (int d = 0; d < DHEAD; ++d) acc += qp[d] * kp[d];
    s = acc * 0.125f;
  }
  S[idx] = s;
}

__global__ void softmax_rows(float* __restrict__ S, int Lk, int rows)
{
  int row = blockIdx.x * (blockDim.x >> 5) + (threadIdx.x >> 5);
  if (row >= rows) return;
  int lane = threadIdx.x & 31;
  float* p = S + (size_t)row * Lk;
  float mx = -INFINITY;
  for (int i = lane; i < Lk; i += 32) mx = fmaxf(mx, p[i]);
  for (int o = 16; o > 0; o >>= 1) mx = fmaxf(mx, __shfl_xor(mx, o, 32));
  float sum = 0.f;
  for (int i = lane; i < Lk; i += 32) { float e = __expf(p[i] - mx); p[i] = e; sum += e; }
  for (int o = 16; o > 0; o >>= 1) sum += __shfl_xor(sum, o, 32);
  float inv = 1.0f / sum;
  for (int i = lane; i < Lk; i += 32) p[i] *= inv;
}

__global__ void attn_av(const float* __restrict__ S, const float* __restrict__ V,
                        float* __restrict__ O, int Lq, int Lk)
{
  int idx = blockIdx.x * blockDim.x + threadIdx.x;
  int total = BATCH * Lq * DMODEL;
  if (idx >= total) return;
  int col = idx & 511;
  int h = col >> 6;
  int q = (idx >> 9) % Lq;
  int b = idx / (DMODEL * Lq);
  const float* sp = S + ((size_t)(b * NHEAD + h) * Lq + q) * Lk;
  const float* vp = V + (size_t)(b * Lk) * DMODEL + col;
  float acc = 0.f;
  for (int k = 0; k < Lk; ++k) acc += sp[k] * vp[(size_t)k * DMODEL];
  O[idx] = acc;
}

// ---------------- layernorm (wave per row, 512 = 32 lanes x 16) ----------------
__global__ void layernorm_rows(const float* __restrict__ X, const float* __restrict__ g,
                               const float* __restrict__ bt, float* __restrict__ Y, int rows)
{
  int row = blockIdx.x * (blockDim.x >> 5) + (threadIdx.x >> 5);
  if (row >= rows) return;
  int lane = threadIdx.x & 31;
  const float* x = X + (size_t)row * DMODEL;
  float vals[16]; float s = 0.f;
#pragma unroll
  for (int i = 0; i < 16; ++i) { vals[i] = x[lane + 32 * i]; s += vals[i]; }
  for (int o = 16; o > 0; o >>= 1) s += __shfl_xor(s, o, 32);
  float mean = s * (1.0f / 512.0f);
  float v = 0.f;
#pragma unroll
  for (int i = 0; i < 16; ++i) { float d = vals[i] - mean; v += d * d; }
  for (int o = 16; o > 0; o >>= 1) v += __shfl_xor(v, o, 32);
  float inv = rsqrtf(v * (1.0f / 512.0f) + 1e-5f);
#pragma unroll
  for (int i = 0; i < 16; ++i) {
    int c = lane + 32 * i;
    Y[(size_t)row * DMODEL + c] = (vals[i] - mean) * inv * g[c] + bt[c];
  }
}

// ---------------- host orchestration ----------------
struct DecP { const float *W1,*W2,*Wk,*Wk2,*Wo,*Wo2,*Wq,*Wq2,*Wv,*Wv2,
                          *b1,*b2,*b3,*bf1,*bf2,*g1,*g2,*g3; };
struct EncP { const float *W1,*W2,*Wk,*Wo,*Wq,*Wv,*b1,*b2,*bf1,*bf2,*g1,*g2; };

extern "C" void kernel_launch(void* const* d_in, const int* in_sizes, int n_in,
                              void* d_out, int out_size, void* d_ws, size_t ws_size,
                              hipStream_t stream)
{
  (void)in_sizes; (void)n_in; (void)out_size; (void)ws_size;
  const float* in_seq   = (const float*)d_in[0];
  const float* in_img   = (const float*)d_in[1];   // (B,64,64,1) == (B,1,64,64), C=1
  const float* in_last  = (const float*)d_in[2];
  const int*   in_sl    = (const int*)d_in[3];
  const int*   in_ll    = (const int*)d_in[4];

  // params: jax tree order (recursively sorted keys)
  int i = 5;
  auto F = [&](int j) { return (const float*)d_in[j]; };
  const float* Wa   = F(i++);   // (49664, 3584)
  const float* Ws_a = F(i++);   // (512, 32)
  const float* Ws_t = F(i++);   // (512, 32)
  const float* Wt   = F(i++);   // (49664, 512)
  const float* ba   = F(i++);   // (3584)
  const float* cb1  = F(i++);  const float* cb2 = F(i++);
  const float* cw1  = F(i++);  const float* cw2 = F(i++);
  const float* fb   = F(i++);  const float* fw  = F(i++);
  const float* bs_a = F(i++);  const float* bs_t = F(i++);
  const float* bt   = F(i++);
  DecP dp[4];
  for (int l = 0; l < 4; ++l) {
    dp[l].W1 = F(i++); dp[l].W2 = F(i++); dp[l].Wk = F(i++); dp[l].Wk2 = F(i++);
    dp[l].Wo = F(i++); dp[l].Wo2 = F(i++); dp[l].Wq = F(i++); dp[l].Wq2 = F(i++);
    dp[l].Wv = F(i++); dp[l].Wv2 = F(i++);
    dp[l].b1 = F(i++); dp[l].b2 = F(i++); dp[l].b3 = F(i++);
    dp[l].bf1 = F(i++); dp[l].bf2 = F(i++);
    dp[l].g1 = F(i++); dp[l].g2 = F(i++); dp[l].g3 = F(i++);
  }
  EncP ep[4];
  for (int l = 0; l < 4; ++l) {
    ep[l].W1 = F(i++); ep[l].W2 = F(i++); ep[l].Wk = F(i++); ep[l].Wo = F(i++);
    ep[l].Wq = F(i++); ep[l].Wv = F(i++);
    ep[l].b1 = F(i++); ep[l].b2 = F(i++); ep[l].bf1 = F(i++); ep[l].bf2 = F(i++);
    ep[l].g1 = F(i++); ep[l].g2 = F(i++);
  }

  // workspace layout
  float* ws = (float*)d_ws;
  size_t off = 0;
  auto alloc = [&](size_t n) { float* p = ws + off; off += n; return p; };
  float* encX  = alloc((size_t)ROWS_E * DMODEL);
  float* decX  = alloc((size_t)ROWS_D * DMODEL);
  float* bufQ  = alloc((size_t)ROWS_E * DMODEL);
  float* bufK  = alloc((size_t)ROWS_E * DMODEL);
  float* bufV  = alloc((size_t)ROWS_E * DMODEL);
  float* bufO  = alloc((size_t)ROWS_E * DMODEL);
  float* bufT  = alloc((size_t)ROWS_E * DMODEL);
  float* bufS  = alloc((size_t)BATCH * NHEAD * LE * LE);
  float* ffnH  = alloc((size_t)ROWS_E * DFFN);
  float* c1    = alloc((size_t)BATCH * 32 * 32 * 32);
  float* c2    = alloc((size_t)BATCH * 64 * 16 * 16);
  float* pool  = alloc((size_t)BATCH * 64);
  float* lvec  = alloc((size_t)BATCH * DMODEL);
  float* headT = alloc((size_t)BATCH * DMODEL);
  float* headA = alloc((size_t)BATCH * 7 * DMODEL);

  auto blocks = [](long long tot) { return (unsigned)((tot + TB - 1) / TB); };
  auto gemm = [&](const float* A, int lda, const float* W, int ldb,
                  const float* bias, const float* resid,
                  float* C, int ldc, int M, int N, int K, int relu) {
    unsigned gb = (unsigned)((M / BM) * (N / BN));
    gemm_bf16_wmma_staged<<<gb, TB, 0, stream>>>(A, lda, W, ldb, bias, resid,
                                                 C, ldc, M, N, K, relu);
  };
  auto gemmS = [&](const float* A, int lda, const float* W, int ldb,
                   const float* bias, float* C, int ldc, int M, int N, int K) {
    int tiles = (M / 16) * (N / 16);
    unsigned gb = (unsigned)((tiles + WPB - 1) / WPB);
    gemm_bf16_wmma_small<<<gb, TB, 0, stream>>>(A, lda, W, ldb, bias, C, ldc, M, N, K);
  };
  auto ln = [&](const float* X, const float* g, const float* b, float* Y, int rows) {
    layernorm_rows<<<blocks((long long)rows * 32), TB, 0, stream>>>(X, g, b, Y, rows);
  };

  // ---- boundary encoder ----
  conv3x3_s2_relu<<<blocks((long long)BATCH*32*32*32), TB, 0, stream>>>(in_img, cw1, cb1, c1, BATCH, 1, 64, 64, 32);
  conv3x3_s2_relu<<<blocks((long long)BATCH*64*16*16), TB, 0, stream>>>(c1, cw2, cb2, c2, BATCH, 32, 32, 32, 64);
  global_pool_16x16<<<blocks(BATCH*64), TB, 0, stream>>>(c2, pool);
  fc64_512<<<blocks(BATCH*DMODEL), TB, 0, stream>>>(pool, fw, fb, lvec);

  // ---- sequence assembly ----
  build_enc_seq<<<blocks((long long)ROWS_E*DMODEL), TB, 0, stream>>>(in_seq, in_sl, lvec, encX);
  build_dec_seq<<<blocks((long long)ROWS_D*DMODEL), TB, 0, stream>>>(in_last, in_ll, decX);

  // ---- encoder layers ----
  for (int l = 0; l < 4; ++l) {
    const EncP& p = ep[l];
    gemm(encX, DMODEL, p.Wq, DMODEL, nullptr, nullptr, bufQ, DMODEL, ROWS_E, DMODEL, DMODEL, 0);
    gemm(encX, DMODEL, p.Wk, DMODEL, nullptr, nullptr, bufK, DMODEL, ROWS_E, DMODEL, DMODEL, 0);
    gemm(encX, DMODEL, p.Wv, DMODEL, nullptr, nullptr, bufV, DMODEL, ROWS_E, DMODEL, DMODEL, 0);
    attn_scores<<<blocks((long long)BATCH*NHEAD*LE*LE), TB, 0, stream>>>(bufQ, bufK, bufS, in_sl, in_ll, LE, LE, 0);
    softmax_rows<<<blocks((long long)BATCH*NHEAD*LE*32), TB, 0, stream>>>(bufS, LE, BATCH*NHEAD*LE);
    attn_av<<<blocks((long long)BATCH*LE*DMODEL), TB, 0, stream>>>(bufS, bufV, bufO, LE, LE);
    gemm(bufO, DMODEL, p.Wo, DMODEL, nullptr, encX, bufT, DMODEL, ROWS_E, DMODEL, DMODEL, 0);
    ln(bufT, p.g1, p.b1, encX, ROWS_E);
    gemm(encX, DMODEL, p.W1, DFFN, p.bf1, nullptr, ffnH, DFFN, ROWS_E, DFFN, DMODEL, 1);
    gemm(ffnH, DFFN, p.W2, DMODEL, p.bf2, encX, bufT, DMODEL, ROWS_E, DMODEL, DFFN, 0);
    ln(bufT, p.g2, p.b2, encX, ROWS_E);
  }

  // ---- decoder layers ----
  for (int l = 0; l < 4; ++l) {
    const DecP& p = dp[l];
    // self-attention (causal)
    gemm(decX, DMODEL, p.Wq, DMODEL, nullptr, nullptr, bufQ, DMODEL, ROWS_D, DMODEL, DMODEL, 0);
    gemm(decX, DMODEL, p.Wk, DMODEL, nullptr, nullptr, bufK, DMODEL, ROWS_D, DMODEL, DMODEL, 0);
    gemm(decX, DMODEL, p.Wv, DMODEL, nullptr, nullptr, bufV, DMODEL, ROWS_D, DMODEL, DMODEL, 0);
    attn_scores<<<blocks((long long)BATCH*NHEAD*LD*LD), TB, 0, stream>>>(bufQ, bufK, bufS, in_sl, in_ll, LD, LD, 1);
    softmax_rows<<<blocks((long long)BATCH*NHEAD*LD*32), TB, 0, stream>>>(bufS, LD, BATCH*NHEAD*LD);
    attn_av<<<blocks((long long)BATCH*LD*DMODEL), TB, 0, stream>>>(bufS, bufV, bufO, LD, LD);
    gemm(bufO, DMODEL, p.Wo, DMODEL, nullptr, decX, bufT, DMODEL, ROWS_D, DMODEL, DMODEL, 0);
    ln(bufT, p.g1, p.b1, decX, ROWS_D);
    // cross-attention (K,V from encoder memory)
    gemm(decX, DMODEL, p.Wq2, DMODEL, nullptr, nullptr, bufQ, DMODEL, ROWS_D, DMODEL, DMODEL, 0);
    gemm(encX, DMODEL, p.Wk2, DMODEL, nullptr, nullptr, bufK, DMODEL, ROWS_E, DMODEL, DMODEL, 0);
    gemm(encX, DMODEL, p.Wv2, DMODEL, nullptr, nullptr, bufV, DMODEL, ROWS_E, DMODEL, DMODEL, 0);
    attn_scores<<<blocks((long long)BATCH*NHEAD*LD*LE), TB, 0, stream>>>(bufQ, bufK, bufS, in_sl, in_ll, LD, LE, 2);
    softmax_rows<<<blocks((long long)BATCH*NHEAD*LD*32), TB, 0, stream>>>(bufS, LE, BATCH*NHEAD*LD);
    attn_av<<<blocks((long long)BATCH*LD*DMODEL), TB, 0, stream>>>(bufS, bufV, bufO, LD, LE);
    gemm(bufO, DMODEL, p.Wo2, DMODEL, nullptr, decX, bufT, DMODEL, ROWS_D, DMODEL, DMODEL, 0);
    ln(bufT, p.g2, p.b2, decX, ROWS_D);
    // FFN
    gemm(decX, DMODEL, p.W1, DFFN, p.bf1, nullptr, ffnH, DFFN, ROWS_D, DFFN, DMODEL, 1);
    gemm(ffnH, DFFN, p.W2, DMODEL, p.bf2, decX, bufT, DMODEL, ROWS_D, DMODEL, DFFN, 0);
    ln(bufT, p.g3, p.b3, decX, ROWS_D);
  }

  // ---- output heads: decX viewed as (32, 97*512 = 49664) ----
  const int KF = LD * DMODEL;
  gemm(decX, KF, Wt, DMODEL, bt, nullptr, headT, DMODEL, BATCH, DMODEL, KF, 0);
  gemm(decX, KF, Wa, 7 * DMODEL, ba, nullptr, headA, 7 * DMODEL, BATCH, 7 * DMODEL, KF, 0);
  // logits -> d_out : [logit_type (32,1,32) | logit_attr (32,7,32)]
  float* out = (float*)d_out;
  gemmS(headT, DMODEL, Ws_t, 32, bs_t, out,        32, BATCH,     32, DMODEL);
  gemmS(headA, DMODEL, Ws_a, 32, bs_a, out + 1024, 32, BATCH * 7, 32, DMODEL);
}
